// EncoderLayer_31765578121430
// MI455X (gfx1250) — matrix operand order repaired and checked by reading, not compiled, not run on hardware
//
#include <hip/hip_runtime.h>

// ---------------------------------------------------------------------------
// Types for CDNA5 WMMA
// ---------------------------------------------------------------------------
typedef __attribute__((ext_vector_type(16))) __bf16 v16bf;
typedef __attribute__((ext_vector_type(8)))  float  v8f;

#define LN_EPS 1e-6f

__device__ __forceinline__ unsigned short f2bf(float f) {
    unsigned int u = __float_as_uint(f);
    unsigned int r = u + 0x7FFFu + ((u >> 16) & 1u);   // round-to-nearest-even
    return (unsigned short)(r >> 16);
}

__device__ __forceinline__ float gelu_exact(float x) {
    return 0.5f * x * (1.0f + erff(x * 0.7071067811865476f));
}

// CDNA5 async global->LDS copy (ASYNCcnt-tracked, bypasses VGPRs).
// ldsaddr = low 32 bits of generic shared pointer (LDS byte address).
__device__ __forceinline__ void async_copy_b128(unsigned ldsaddr, const void* gaddr) {
    asm volatile("global_load_async_to_lds_b128 %0, %1, off"
                 :: "v"(ldsaddr), "v"((unsigned long long)(size_t)gaddr)
                 : "memory");
}
__device__ __forceinline__ void wait_async0() {
    asm volatile("s_wait_asynccnt 0" ::: "memory");
}

// ---------------------------------------------------------------------------
// fp32 -> bf16 conversion
// ---------------------------------------------------------------------------
__global__ __launch_bounds__(256) void cvt_f32_to_bf16(
    const float* __restrict__ src, unsigned short* __restrict__ dst, long n)
{
    long i = (long)blockIdx.x * 256 + threadIdx.x;
    long stride = (long)gridDim.x * 256;
    for (; i < n; i += stride) dst[i] = f2bf(src[i]);
}

// ---------------------------------------------------------------------------
// Generic tiled bf16 WMMA GEMM:  C = act(alpha * A*B + bias)
//   block tile 128x64, 8 waves (4x2 grid), each wave 32x32 (2x2 wmma tiles)
//   K-step 32; LDS staging via async global->LDS where layout permits.
//   - Af != nullptr: A is fp32, converted to bf16 during staging (VALU path)
//   - transB: B element (k,n) is at B[n*ldb + k]  (async path, contiguous k)
//   - batched via blockIdx.z: off = (z/Hdim)*s?b + (z%Hdim)*s?h
// ---------------------------------------------------------------------------
#define BM 128
#define BN 64
#define BKT 32
#define AST 40   // LDS row stride for As (halves)
#define BST 40   // LDS row stride for Bs (halves)

__global__ __launch_bounds__(256) void gemm_bf16_wmma(
    const unsigned short* __restrict__ A, const float* __restrict__ Af,
    long lda, long sAb, long sAh,
    const unsigned short* __restrict__ Bm, long ldb, long sBb, long sBh, int transB,
    float* __restrict__ Cf, long ldcf, long sCfb, long sCfh,
    unsigned short* __restrict__ Cb, long ldcb, long sCbb, long sCbh,
    int Kdim, int Hdim,
    const float* __restrict__ bias, float alpha, int act)
{
    __shared__ unsigned short As[BM * AST];
    __shared__ unsigned short Bs[BN * BST];

    const int tid   = threadIdx.x;
    const int lane  = tid & 31;
    const int wave  = tid >> 5;
    const int waveM = wave >> 1;      // 0..3
    const int waveN = wave & 1;       // 0..1
    const int half  = lane >> 4;      // 0..1
    const int l16   = lane & 15;      // 0..15

    const long z  = blockIdx.z;
    const long zb = z / Hdim;
    const long zh = z % Hdim;
    const long offA  = zb * sAb  + zh * sAh;
    const long offB  = zb * sBb  + zh * sBh;
    const long offCf = zb * sCfb + zh * sCfh;
    const long offCb = zb * sCbb + zh * sCbh;
    const long m0 = (long)blockIdx.y * BM;
    const long n0 = (long)blockIdx.x * BN;

    v8f acc[2][2];
    const v8f vzero = {0.f, 0.f, 0.f, 0.f, 0.f, 0.f, 0.f, 0.f};
    acc[0][0] = vzero; acc[0][1] = vzero; acc[1][0] = vzero; acc[1][1] = vzero;

    const bool anyAsync = (Af == nullptr) || transB;

    const int nk = Kdim / BKT;
    for (int ko = 0; ko < nk; ++ko) {
        const long kbase = (long)ko * BKT;

        // ---- stage A tile (128 x 32 bf16) : 512 chunks of 8 halves ----
        if (Af == nullptr) {
            // async global -> LDS, no VGPR round trip
#pragma unroll
            for (int it = 0; it < 2; ++it) {
                int c    = tid + it * 256;
                int row  = c >> 2;
                int col8 = (c & 3) * 8;
                const unsigned short* gp = A + offA + (m0 + row) * lda + kbase + col8;
                unsigned ldsaddr = (unsigned)(size_t)(const void*)&As[row * AST + col8];
                async_copy_b128(ldsaddr, gp);
            }
        } else {
#pragma unroll
            for (int it = 0; it < 2; ++it) {
                int c    = tid + it * 256;
                int row  = c >> 2;
                int col8 = (c & 3) * 8;
                const float4* gp = reinterpret_cast<const float4*>(
                    Af + offA + (m0 + row) * lda + kbase + col8);
                float4 d0 = gp[0], d1 = gp[1];
                unsigned int* lp = reinterpret_cast<unsigned int*>(&As[row * AST + col8]);
                lp[0] = (unsigned)f2bf(d0.x) | ((unsigned)f2bf(d0.y) << 16);
                lp[1] = (unsigned)f2bf(d0.z) | ((unsigned)f2bf(d0.w) << 16);
                lp[2] = (unsigned)f2bf(d1.x) | ((unsigned)f2bf(d1.y) << 16);
                lp[3] = (unsigned)f2bf(d1.z) | ((unsigned)f2bf(d1.w) << 16);
            }
        }

        // ---- stage B tile (32 x 64 bf16), stored transposed: Bs[n][k] ----
        if (!transB) {
            // source contiguous along n, LDS wants [n][k]: 16-bit scatter (VALU)
            int k   = tid >> 3;          // 0..31
            int nn8 = (tid & 7) * 8;     // 0..56
            const uint4* gp = reinterpret_cast<const uint4*>(
                Bm + offB + (kbase + k) * ldb + n0 + nn8);
            uint4 d = *gp;
            unsigned int w[4] = {d.x, d.y, d.z, d.w};
#pragma unroll
            for (int j = 0; j < 8; ++j)
                Bs[(nn8 + j) * BST + k] = (unsigned short)(w[j >> 1] >> ((j & 1) * 16));
        } else {
            // source contiguous along k == LDS layout: async copy
            int nn = tid >> 2;           // 0..63
            int k8 = (tid & 3) * 8;      // 0..24
            const unsigned short* gp = Bm + offB + (n0 + nn) * ldb + kbase + k8;
            unsigned ldsaddr = (unsigned)(size_t)(const void*)&Bs[nn * BST + k8];
            async_copy_b128(ldsaddr, gp);
        }

        if (anyAsync) wait_async0();
        __syncthreads();

        // ---- load fragments per ISA 7.12.2 layouts, issue 4 WMMAs ----
        union Frag { unsigned int u[8]; v16bf v; } afr[2], bfr[2];
#pragma unroll
        for (int t = 0; t < 2; ++t) {
            int m = waveM * 32 + t * 16 + l16;
#pragma unroll
            for (int r = 0; r < 8; ++r) {
                int k2 = (r < 4 ? 2 * r : 16 + 2 * (r - 4)) + 8 * half;
                afr[t].u[r] = *reinterpret_cast<const unsigned int*>(&As[m * AST + k2]);
            }
            int n = waveN * 32 + t * 16 + l16;
#pragma unroll
            for (int r = 0; r < 8; ++r) {
                int k2 = (r < 4 ? 2 * r : 16 + 2 * (r - 4)) + 8 * half;
                bfr[t].u[r] = *reinterpret_cast<const unsigned int*>(&Bs[n * BST + k2]);
            }
        }
#pragma unroll
        for (int t = 0; t < 2; ++t)
#pragma unroll
            for (int u = 0; u < 2; ++u)
                acc[t][u] = __builtin_amdgcn_wmma_f32_16x16x32_bf16(
                    false, afr[t].v, false, bfr[u].v,
                    (short)0, acc[t][u], false, false);

        __syncthreads();
    }

    // ---- epilogue: alpha, bias, activation, f32 and/or bf16 stores ----
#pragma unroll
    for (int t = 0; t < 2; ++t) {
#pragma unroll
        for (int u = 0; u < 2; ++u) {
            long n = n0 + waveN * 32 + u * 16 + l16;
            float bv = bias ? bias[n] : 0.0f;
#pragma unroll
            for (int r = 0; r < 8; ++r) {
                long m = m0 + waveM * 32 + t * 16 + r + 8 * half;
                float val = acc[t][u][r] * alpha + bv;
                if (act) val = gelu_exact(val);
                if (Cf) Cf[offCf + m * ldcf + n] = val;
                if (Cb) Cb[offCb + m * ldcb + n] = f2bf(val);
            }
        }
    }
}

// ---------------------------------------------------------------------------
// In-place masked softmax over rows of attn_prob [B,H,S,S] (S = 1024)
// ---------------------------------------------------------------------------
__global__ __launch_bounds__(256) void softmax_mask(
    float* __restrict__ P, const unsigned char* __restrict__ mask)
{
    __shared__ float red[256];
    const long rid = blockIdx.x;            // (b*H + h)*S + sq
    const int  tid = threadIdx.x;
    const long b  = rid >> 14;              // / (H*S) = 16384
    const long sq = rid & 1023;
    float* p = P + rid * 1024L;
    const unsigned char* mrow = mask + (((b << 10) + sq) << 10);

    float v[4];
    float mx = -3.4e38f;
#pragma unroll
    for (int i = 0; i < 4; ++i) {
        int k = tid + i * 256;
        float s = p[k];
        if (mrow[k]) s = -1e9f;
        v[i] = s;
        mx = fmaxf(mx, s);
    }
    red[tid] = mx; __syncthreads();
    for (int s2 = 128; s2 > 0; s2 >>= 1) {
        if (tid < s2) red[tid] = fmaxf(red[tid], red[tid + s2]);
        __syncthreads();
    }
    mx = red[0]; __syncthreads();

    float sum = 0.0f;
#pragma unroll
    for (int i = 0; i < 4; ++i) { v[i] = expf(v[i] - mx); sum += v[i]; }
    red[tid] = sum; __syncthreads();
    for (int s2 = 128; s2 > 0; s2 >>= 1) {
        if (tid < s2) red[tid] += red[tid + s2];
        __syncthreads();
    }
    float inv = 1.0f / red[0];
#pragma unroll
    for (int i = 0; i < 4; ++i) p[tid + i * 256] = v[i] * inv;
}

// ---------------------------------------------------------------------------
// LayerNorm over last dim (1024), optional residual add: Y = LN(X (+R))*g + b
// ---------------------------------------------------------------------------
__global__ __launch_bounds__(256) void layernorm_kernel(
    const float* __restrict__ X, const float* __restrict__ R,
    const float* __restrict__ g, const float* __restrict__ bt,
    float* __restrict__ Y)
{
    __shared__ float red[256];
    const long row = blockIdx.x;
    const int  tid = threadIdx.x;
    const float* x = X + row * 1024L;
    const float* r = R ? R + row * 1024L : nullptr;

    float v[4];
    float s = 0.0f;
#pragma unroll
    for (int i = 0; i < 4; ++i) {
        int k = tid + i * 256;
        float t = x[k];
        if (r) t += r[k];
        v[i] = t;
        s += t;
    }
    red[tid] = s; __syncthreads();
    for (int s2 = 128; s2 > 0; s2 >>= 1) {
        if (tid < s2) red[tid] += red[tid + s2];
        __syncthreads();
    }
    float mu = red[0] * (1.0f / 1024.0f); __syncthreads();

    float vs = 0.0f;
#pragma unroll
    for (int i = 0; i < 4; ++i) { float d = v[i] - mu; vs += d * d; }
    red[tid] = vs; __syncthreads();
    for (int s2 = 128; s2 > 0; s2 >>= 1) {
        if (tid < s2) red[tid] += red[tid + s2];
        __syncthreads();
    }
    float rstd = rsqrtf(red[0] * (1.0f / 1024.0f) + LN_EPS);
#pragma unroll
    for (int i = 0; i < 4; ++i) {
        int k = tid + i * 256;
        Y[row * 1024L + k] = (v[i] - mu) * rstd * g[k] + bt[k];
    }
}

// ---------------------------------------------------------------------------
// Host launcher
// ---------------------------------------------------------------------------
static void launch_gemm(hipStream_t stream,
    const unsigned short* A, const float* Af, long lda, long sAb, long sAh,
    const unsigned short* Bm, long ldb, long sBb, long sBh, int transB,
    float* Cf, long ldcf, long sCfb, long sCfh,
    unsigned short* Cb, long ldcb, long sCbb, long sCbh,
    int Mdim, int Ndim, int Kdim, int Z, int Hdim,
    const float* bias, float alpha, int act)
{
    dim3 g(Ndim / BN, Mdim / BM, Z);
    gemm_bf16_wmma<<<g, 256, 0, stream>>>(A, Af, lda, sAb, sAh,
        Bm, ldb, sBb, sBh, transB,
        Cf, ldcf, sCfb, sCfh, Cb, ldcb, sCbb, sCbh,
        Kdim, Hdim, bias, alpha, act);
}

static void launch_cvt(hipStream_t stream, const float* src, unsigned short* dst, long n)
{
    long blocks = (n + 255) / 256;
    if (blocks > 16384) blocks = 16384;
    cvt_f32_to_bf16<<<(unsigned)blocks, 256, 0, stream>>>(src, dst, n);
}

extern "C" void kernel_launch(void* const* d_in, const int* in_sizes, int n_in,
                              void* d_out, int out_size, void* d_ws, size_t ws_size,
                              hipStream_t stream)
{
    (void)in_sizes; (void)n_in; (void)out_size; (void)ws_size;

    constexpr int  Bc = 8, Sc = 1024, DHc = 1024, Hc = 16, Dc = 64, DFc = 4096;
    constexpr long Mrows = (long)Bc * Sc;     // 8192

    const float*         inX  = (const float*)d_in[0];
    const unsigned char* mask = (const unsigned char*)d_in[1];
    const float* Wq = (const float*)d_in[2];  const float* bq = (const float*)d_in[3];
    const float* Wk = (const float*)d_in[4];  const float* bk = (const float*)d_in[5];
    const float* Wv = (const float*)d_in[6];  const float* bv = (const float*)d_in[7];
    const float* Wo = (const float*)d_in[8];  const float* bo = (const float*)d_in[9];
    const float* W1 = (const float*)d_in[10]; const float* b1 = (const float*)d_in[11];
    const float* W2 = (const float*)d_in[12]; const float* b2 = (const float*)d_in[13];
    const float* g1 = (const float*)d_in[14]; const float* be1 = (const float*)d_in[15];
    const float* g2 = (const float*)d_in[16]; const float* be2 = (const float*)d_in[17];

    float* out  = (float*)d_out;
    float* attn = out + Mrows * DHc;          // attn_prob region [B,H,S,S] fp32

    // -------- workspace carve-out --------
    char*  ws  = (char*)d_ws;
    size_t off = 0;
    auto alloc = [&](size_t bytes) -> char* {
        char* p = ws + off;
        off = (off + bytes + 255) & ~(size_t)255;
        return p;
    };
    unsigned short* Xb    = (unsigned short*)alloc(Mrows * DHc * 2);
    unsigned short* Wqb   = (unsigned short*)alloc((long)DHc * DHc * 2);
    unsigned short* Wkb   = (unsigned short*)alloc((long)DHc * DHc * 2);
    unsigned short* Wvb   = (unsigned short*)alloc((long)DHc * DHc * 2);
    unsigned short* Wob   = (unsigned short*)alloc((long)DHc * DHc * 2);
    unsigned short* W1b   = (unsigned short*)alloc((long)DHc * DFc * 2);
    unsigned short* W2b   = (unsigned short*)alloc((long)DFc * DHc * 2);
    unsigned short* Qb    = (unsigned short*)alloc(Mrows * DHc * 2);
    unsigned short* Kb    = (unsigned short*)alloc(Mrows * DHc * 2);
    unsigned short* Vb    = (unsigned short*)alloc(Mrows * DHc * 2);
    unsigned short* ffn1b = (unsigned short*)alloc(Mrows * DFc * 2);
    float*          attR  = (float*)alloc(Mrows * DHc * 4);  // pre-LN1; reused as ffn2
    float*          attO  = (float*)alloc(Mrows * DHc * 4);  // LN1 output
    unsigned short* ctxb  = Qb;                              // Q dead after scores
    float*          ffn2  = attR;                            // attR dead after LN1

    // -------- 1. fp32 -> bf16 conversions --------
    launch_cvt(stream, inX, Xb, Mrows * DHc);
    launch_cvt(stream, Wq, Wqb, (long)DHc * DHc);
    launch_cvt(stream, Wk, Wkb, (long)DHc * DHc);
    launch_cvt(stream, Wv, Wvb, (long)DHc * DHc);
    launch_cvt(stream, Wo, Wob, (long)DHc * DHc);
    launch_cvt(stream, W1, W1b, (long)DHc * DFc);
    launch_cvt(stream, W2, W2b, (long)DFc * DHc);

    // -------- 2. QKV projections: [8192,1024] @ [1024,1024] + bias --------
    launch_gemm(stream, Xb, nullptr, DHc, 0, 0, Wqb, DHc, 0, 0, 0,
                nullptr, 0, 0, 0, Qb, DHc, 0, 0,
                (int)Mrows, DHc, DHc, 1, 1, bq, 1.0f, 0);
    launch_gemm(stream, Xb, nullptr, DHc, 0, 0, Wkb, DHc, 0, 0, 0,
                nullptr, 0, 0, 0, Kb, DHc, 0, 0,
                (int)Mrows, DHc, DHc, 1, 1, bk, 1.0f, 0);
    launch_gemm(stream, Xb, nullptr, DHc, 0, 0, Wvb, DHc, 0, 0, 0,
                nullptr, 0, 0, 0, Vb, DHc, 0, 0,
                (int)Mrows, DHc, DHc, 1, 1, bv, 1.0f, 0);

    // -------- 3. scores = Q K^T * 1/sqrt(D), per (b,h), into attn region ----
    launch_gemm(stream, Qb, nullptr, DHc, (long)Sc * DHc, Dc,
                Kb, DHc, (long)Sc * DHc, Dc, 1,
                attn, Sc, (long)Hc * Sc * Sc, (long)Sc * Sc,
                nullptr, 0, 0, 0,
                Sc, Sc, Dc, Bc * Hc, Hc, nullptr, 0.125f, 0);

    // -------- 4. masked softmax in-place on attn --------
    softmax_mask<<<Bc * Hc * Sc, 256, 0, stream>>>(attn, mask);

    // -------- 5. context = P V (A fp32 converted on the fly) --------
    launch_gemm(stream, nullptr, attn, Sc, (long)Hc * Sc * Sc, (long)Sc * Sc,
                Vb, DHc, (long)Sc * DHc, Dc, 0,
                nullptr, 0, 0, 0,
                ctxb, DHc, (long)Sc * DHc, Dc,
                Sc, Dc, Sc, Bc * Hc, Hc, nullptr, 1.0f, 0);

    // -------- 6. att_raw = ctx @ Wo + bo --------
    launch_gemm(stream, ctxb, nullptr, DHc, 0, 0, Wob, DHc, 0, 0, 0,
                attR, DHc, 0, 0, nullptr, 0, 0, 0,
                (int)Mrows, DHc, DHc, 1, 1, bo, 1.0f, 0);

    // -------- 7. LN1 --------
    layernorm_kernel<<<(unsigned)Mrows, 256, 0, stream>>>(attR, nullptr, g1, be1, attO);

    // -------- 8. ffn1 = gelu(X @ W1 + b1) --------
    launch_gemm(stream, Xb, nullptr, DHc, 0, 0, W1b, DFc, 0, 0, 0,
                nullptr, 0, 0, 0, ffn1b, DFc, 0, 0,
                (int)Mrows, DFc, DHc, 1, 1, b1, 1.0f, 1);

    // -------- 9. ffn2 = ffn1 @ W2 + b2 --------
    launch_gemm(stream, ffn1b, nullptr, DFc, 0, 0, W2b, DHc, 0, 0, 0,
                ffn2, DHc, 0, 0, nullptr, 0, 0, 0,
                (int)Mrows, DHc, DFc, 1, 1, b2, 1.0f, 0);

    // -------- 10. out = LN2(ffn2 + att_out) --------
    layernorm_kernel<<<(unsigned)Mrows, 256, 0, stream>>>(ffn2, attO, g2, be2, out);
}
